// Contrastive_Model_47004122087946
// MI455X (gfx1250) — matrix operand order repaired and checked by reading, not compiled, not run on hardware
//
#include <hip/hip_runtime.h>
#include <hip/hip_bf16.h>

#define DIM   128
#define NCSV  8192
#define NWIKI 8192
#define INV_T 10.0f   // 1/temperature

typedef __attribute__((ext_vector_type(16))) __bf16    v16bf;
typedef __attribute__((ext_vector_type(8)))  __bf16    v8bf;
typedef __attribute__((ext_vector_type(8)))  float     v8f;
typedef __attribute__((ext_vector_type(4)))  unsigned  v4u;
typedef __attribute__((ext_vector_type(8)))  int       v8i;
typedef __attribute__((ext_vector_type(4)))  int       v4i;

#if defined(__has_builtin)
#  if __has_builtin(__builtin_amdgcn_tensor_load_to_lds)
#    define HAVE_TDM 1
#  endif
#endif

// LDS tile: 128 rows, row stride 136 bf16 (= 272 B = 68 DWORDs) so the TDM pad
// (4 DWORDs every 64 payload DWORDs) skews rows across the 64 LDS banks.
#define LDS_STRIDE 136

// ---------------- workspace layout (4-byte words) ----------------
static constexpr size_t CD          = (size_t)NCSV * DIM;            // 1,048,576
static constexpr size_t OFF_CSVSUM  = 0;
static constexpr size_t OFF_WIKISUM = OFF_CSVSUM + CD;
static constexpr size_t OFF_CSVCNT  = OFF_WIKISUM + CD;
static constexpr size_t OFF_WIKICNT = OFF_CSVCNT + NCSV;
static constexpr size_t OFF_ROWSE   = OFF_WIKICNT + NWIKI;
static constexpr size_t OFF_COLSE   = OFF_ROWSE + NCSV;
static constexpr size_t OFF_ROWLOSS = OFF_COLSE + NWIKI;
static constexpr size_t OFF_ROWCNT  = OFF_ROWLOSS + NCSV;
static constexpr size_t OFF_COLLOSS = OFF_ROWCNT + NCSV;
static constexpr size_t OFF_COLCNT  = OFF_COLLOSS + NWIKI;
static constexpr size_t OFF_BITMAP  = OFF_COLCNT + NWIKI;            // C*W/32 words
static constexpr size_t BITWORDS    = (size_t)NCSV * NWIKI / 32;     // 2,097,152
static constexpr size_t OFF_CSVBF   = OFF_BITMAP + BITWORDS;         // bf16: CD/2 words
static constexpr size_t OFF_WIKIBF  = OFF_CSVBF + CD / 2;
static constexpr size_t ZERO_WORDS  = OFF_CSVBF;                     // everything before bf16 bufs

// ---------------- kernels ----------------
__global__ void k_zero(unsigned* p, unsigned n) {
    unsigned i = blockIdx.x * blockDim.x + threadIdx.x;
    if (i < n) p[i] = 0u;
}

// segment sums + counts via hardware f32 atomics
__global__ void k_segsum(const float* __restrict__ f1, const float* __restrict__ f2,
                         const int* __restrict__ cid, const int* __restrict__ wid,
                         float* __restrict__ csvSum, float* __restrict__ wikiSum,
                         float* __restrict__ csvCnt, float* __restrict__ wikiCnt, int n) {
    int tid = blockIdx.x * blockDim.x + threadIdx.x;
    int i = tid >> 5;           // example index
    int ch = tid & 31;          // float4 chunk (32 * 4 = 128 dims)
    if (i >= n) return;
    int c = cid[i], w = wid[i];
    float4 a = ((const float4*)(f1 + (size_t)i * DIM))[ch];
    float4 b = ((const float4*)(f2 + (size_t)i * DIM))[ch];
    float* cs = csvSum + (size_t)c * DIM + ch * 4;
    float* wsm = wikiSum + (size_t)w * DIM + ch * 4;
    unsafeAtomicAdd(cs + 0, a.x); unsafeAtomicAdd(cs + 1, a.y);
    unsafeAtomicAdd(cs + 2, a.z); unsafeAtomicAdd(cs + 3, a.w);
    unsafeAtomicAdd(wsm + 0, b.x); unsafeAtomicAdd(wsm + 1, b.y);
    unsafeAtomicAdd(wsm + 2, b.z); unsafeAtomicAdd(wsm + 3, b.w);
    if (ch == 0) { unsafeAtomicAdd(csvCnt + c, 1.0f); unsafeAtomicAdd(wikiCnt + w, 1.0f); }
}

// sums -> means (in place) + bf16 copies for WMMA
__global__ void k_finalize(float* __restrict__ csvSum, float* __restrict__ wikiSum,
                           const float* __restrict__ csvCnt, const float* __restrict__ wikiCnt,
                           __bf16* __restrict__ csvB, __bf16* __restrict__ wikiB) {
    unsigned idx = blockIdx.x * blockDim.x + threadIdx.x;
    if (idx >= CD) return;
    unsigned row = idx >> 7;  // /DIM
    float cv = csvSum[idx] / fmaxf(csvCnt[row], 1.0f);
    float wv = wikiSum[idx] / fmaxf(wikiCnt[row], 1.0f);
    csvSum[idx] = cv;  wikiSum[idx] = wv;
    csvB[idx] = (__bf16)cv;  wikiB[idx] = (__bf16)wv;
}

// WMMA GEMM tilewise; B tile staged to LDS by the Tensor Data Mover;
// accumulate exp(logit) into row/col sum-exp (logits never stored).
__global__ __launch_bounds__(256)
void k_logits_sumexp(const __bf16* __restrict__ csvB, const __bf16* __restrict__ wikiB,
                     float* __restrict__ rowSE, float* __restrict__ colSE) {
    __shared__ __bf16 tileB[128 * LDS_STRIDE];   // 34,816 B (padded rows)

    const int lane = threadIdx.x & 31;
    const int wv   = threadIdx.x >> 5;          // 8 waves
    const int hi   = lane >> 4;                 // half-wave selector
    const int l15  = lane & 15;
    const int c0   = blockIdx.x * 128 + wv * 16;  // 16 rows per wave
    const int w0   = blockIdx.y * 128;            // 128 cols per block

#if defined(HAVE_TDM)
    if (threadIdx.x < 32) {   // wave 0 issues one TDM descriptor for the whole block
        unsigned long long ga = (unsigned long long)(const void*)(wikiB + (size_t)w0 * DIM);
        unsigned ldsAddr = (unsigned)(__SIZE_TYPE__)&tileB[0];  // LDS byte offset (low 32 bits)
        // D# group 0: count=1 | lds_addr | global_addr(57b) | type=2
        v4u g0;
        g0.x = 1u;
        g0.y = ldsAddr;
        g0.z = (unsigned)(ga & 0xFFFFFFFFull);
        g0.w = (unsigned)((ga >> 32) & 0x01FFFFFFull) | (2u << 30);
        // D# group 1: data_size=2B, pad 4 DW every 64 DW, tensor 128 x 8192 (stride 128),
        //             tile 128 x 128
        v8i g1;
        g1.s0 = (int)((1u << 16) | (1u << 20) | (5u << 22) | (3u << 25));
        g1.s1 = (int)(128u << 16);    // tensor_dim0[15:0]=128 in bits 79:64... (bits 63:48)
        g1.s2 = (int)(8192u << 16);   // tensor_dim0 hi=0 | tensor_dim1 lo=8192
        g1.s3 = (int)(128u << 16);    // tensor_dim1 hi=0 | tile_dim0=128
        g1.s4 = (int)(128u);          // tile_dim1=128 | tile_dim2=0
        g1.s5 = (int)(128u);          // tensor_dim0_stride[31:0]=128
        g1.s6 = 0;                    // stride hi | tensor_dim1_stride lo
        g1.s7 = 0;
        v4i zg = {0, 0, 0, 0};        // groups 2/3 unused (2-D tensor)
#  if __clang_major__ >= 23
        v8i zg4 = {0, 0, 0, 0, 0, 0, 0, 0};
        __builtin_amdgcn_tensor_load_to_lds(g0, g1, zg, zg, zg4, 0);
#  else
        __builtin_amdgcn_tensor_load_to_lds(g0, g1, zg, zg, 0);
#  endif
        __builtin_amdgcn_s_wait_tensorcnt((short)0);
    }
    __syncthreads();
#else
    // Fallback: cooperative staging with the same padded layout
    for (int idx = threadIdx.x; idx < 128 * 16; idx += 256) {
        int row = idx >> 4, seg = idx & 15;
        *(v8bf*)&tileB[row * LDS_STRIDE + seg * 8] =
            *(const v8bf*)(wikiB + (size_t)(w0 + row) * DIM + seg * 8);
    }
    __syncthreads();
#endif

    v8f acc[8];
    #pragma unroll
    for (int t = 0; t < 8; ++t)
        #pragma unroll
        for (int j = 0; j < 8; ++j) acc[t][j] = 0.0f;

    // A fragment base: row = c0 + (lane&15); 16 contiguous K per half
    const __bf16* aBase = csvB + (size_t)(c0 + l15) * DIM + hi * 16;

    #pragma unroll
    for (int kk = 0; kk < DIM; kk += 32) {
        v8bf alo = *(const v8bf*)(aBase + kk);        // K = kk+kb   .. +7
        v8bf ahi = *(const v8bf*)(aBase + kk + 8);    // K = kk+kb+8 .. +15
        v16bf a = __builtin_shufflevector(alo, ahi, 0,1,2,3,4,5,6,7,8,9,10,11,12,13,14,15);
        #pragma unroll
        for (int t = 0; t < 8; ++t) {
            // B fragment from LDS: column (t*16 + lane&15), 16 contiguous K per half
            const __bf16* bp = &tileB[(t * 16 + l15) * LDS_STRIDE + kk + hi * 16];
            v16bf b = *(const v16bf*)bp;
            acc[t] = __builtin_amdgcn_wmma_f32_16x16x32_bf16(
                false, a, false, b, (short)0, acc[t], false, false);
        }
    }

    // C layout: element j of lane L -> row (j + 8*hi), col (L&15)
    float rAcc[8];
    #pragma unroll
    for (int j = 0; j < 8; ++j) rAcc[j] = 0.0f;

    #pragma unroll
    for (int t = 0; t < 8; ++t) {
        float colPart = 0.0f;
        #pragma unroll
        for (int j = 0; j < 8; ++j) {
            float v = __expf(acc[t][j] * INV_T);
            colPart += v;
            float r = v;    // reduce across the 16 lanes holding this row
            r += __shfl_xor(r, 1);  r += __shfl_xor(r, 2);
            r += __shfl_xor(r, 4);  r += __shfl_xor(r, 8);
            rAcc[j] += r;
        }
        colPart += __shfl_xor(colPart, 16);  // combine the two 8-row halves
        if (lane < 16) unsafeAtomicAdd(&colSE[w0 + t * 16 + l15], colPart);
    }
    #pragma unroll
    for (int j = 0; j < 8; ++j)
        if (l15 == 0) unsafeAtomicAdd(&rowSE[c0 + hi * 8 + j], rAcc[j]);
}

// dedup pairs via bitmap atomicOr; owner recomputes f32 logit, accumulates log-softmax terms
__global__ void k_pairs(const int* __restrict__ cid, const int* __restrict__ wid,
                        const float* __restrict__ csvE, const float* __restrict__ wikiE,
                        const float* __restrict__ rowSE, const float* __restrict__ colSE,
                        unsigned* __restrict__ bitmap,
                        float* __restrict__ rowLoss, float* __restrict__ rowCnt,
                        float* __restrict__ colLoss, float* __restrict__ colCnt, int n) {
    int i = blockIdx.x * blockDim.x + threadIdx.x;
    if (i >= n) return;
    int c = cid[i], w = wid[i];
    unsigned bitIdx = (unsigned)c * NWIKI + (unsigned)w;
    unsigned old = atomicOr(&bitmap[bitIdx >> 5], 1u << (bitIdx & 31));
    if (old & (1u << (bitIdx & 31))) return;   // duplicate pair
    const float4* ca = (const float4*)(csvE + (size_t)c * DIM);
    const float4* wa = (const float4*)(wikiE + (size_t)w * DIM);
    float dot = 0.0f;
    #pragma unroll 4
    for (int q = 0; q < DIM / 4; ++q) {
        float4 x = ca[q], y = wa[q];
        dot += x.x * y.x + x.y * y.y + x.z * y.z + x.w * y.w;
    }
    float logit = dot * INV_T;
    unsafeAtomicAdd(&rowLoss[c], logit - __logf(rowSE[c]));
    unsafeAtomicAdd(&rowCnt[c], 1.0f);
    unsafeAtomicAdd(&colLoss[w], logit - __logf(colSE[w]));
    unsafeAtomicAdd(&colCnt[w], 1.0f);
}

__global__ void k_final(const float* __restrict__ rowLoss, const float* __restrict__ rowCnt,
                        const float* __restrict__ colLoss, const float* __restrict__ colCnt,
                        float* __restrict__ out) {
    __shared__ float s1[256], s0[256];
    float r = 0.0f, cl = 0.0f;
    for (int i = threadIdx.x; i < NCSV; i += 256)
        r += rowLoss[i] / fmaxf(rowCnt[i], 1.0f);
    for (int i = threadIdx.x; i < NWIKI; i += 256)
        cl += colLoss[i] / fmaxf(colCnt[i], 1.0f);
    s1[threadIdx.x] = r; s0[threadIdx.x] = cl;
    __syncthreads();
    for (int s = 128; s > 0; s >>= 1) {
        if (threadIdx.x < s) { s1[threadIdx.x] += s1[threadIdx.x + s]; s0[threadIdx.x] += s0[threadIdx.x + s]; }
        __syncthreads();
    }
    if (threadIdx.x == 0) {
        float axis1 = -s1[0] / (float)NCSV;   // A1 = 0.5
        float axis0 = -s0[0] / (float)NWIKI;  // A0 = 0.5
        out[0] = 0.5f * axis0 + 0.5f * axis1;
    }
}

// ---------------- launch ----------------
extern "C" void kernel_launch(void* const* d_in, const int* in_sizes, int n_in,
                              void* d_out, int out_size, void* d_ws, size_t ws_size,
                              hipStream_t stream) {
    const float* f1 = (const float*)d_in[0];
    const float* f2 = (const float*)d_in[1];
    const int* cid = (const int*)d_in[2];
    const int* wid = (const int*)d_in[3];
    const int N = in_sizes[0] / DIM;

    float* ws = (float*)d_ws;
    float* csvSum  = ws + OFF_CSVSUM;
    float* wikiSum = ws + OFF_WIKISUM;
    float* csvCnt  = ws + OFF_CSVCNT;
    float* wikiCnt = ws + OFF_WIKICNT;
    float* rowSE   = ws + OFF_ROWSE;
    float* colSE   = ws + OFF_COLSE;
    float* rowLoss = ws + OFF_ROWLOSS;
    float* rowCnt  = ws + OFF_ROWCNT;
    float* colLoss = ws + OFF_COLLOSS;
    float* colCnt  = ws + OFF_COLCNT;
    unsigned* bitmap = (unsigned*)(ws + OFF_BITMAP);
    __bf16* csvB  = (__bf16*)(ws + OFF_CSVBF);
    __bf16* wikiB = (__bf16*)(ws + OFF_WIKIBF);

    k_zero<<<(unsigned)((ZERO_WORDS + 255) / 256), 256, 0, stream>>>((unsigned*)ws, (unsigned)ZERO_WORDS);

    k_segsum<<<(N * 32 + 255) / 256, 256, 0, stream>>>(f1, f2, cid, wid,
                                                       csvSum, wikiSum, csvCnt, wikiCnt, N);

    k_finalize<<<(unsigned)((CD + 255) / 256), 256, 0, stream>>>(csvSum, wikiSum, csvCnt, wikiCnt,
                                                                 csvB, wikiB);

    dim3 g(NCSV / 128, NWIKI / 128);
    k_logits_sumexp<<<g, 256, 0, stream>>>(csvB, wikiB, rowSE, colSE);

    k_pairs<<<(N + 255) / 256, 256, 0, stream>>>(cid, wid, csvSum, wikiSum, rowSE, colSE,
                                                 bitmap, rowLoss, rowCnt, colLoss, colCnt, N);

    k_final<<<1, 256, 0, stream>>>(rowLoss, rowCnt, colLoss, colCnt, (float*)d_out);
}
